// Decoder_18107582120580
// MI455X (gfx1250) — matrix operand order repaired and checked by reading, not compiled, not run on hardware
//
#include <hip/hip_runtime.h>
#include <hip/hip_bf16.h>
#include <math.h>

typedef __attribute__((ext_vector_type(16))) _Float16 v16h;
typedef __attribute__((ext_vector_type(8)))  _Float16 v8h;
typedef __attribute__((ext_vector_type(8)))  float    v8f;

#define NWG   32
#define NTHR  256
#define NTOT  (NWG*NTHR)        // 8192 threads
#define NWAVE (NWG*(NTHR/32))   // 256 waves (wave32)

static constexpr int CB   = 64;    // batch
static constexpr int CT   = 256;   // T_ENC
static constexpr int CIN  = 256;   // IN_FEAT
static constexpr int CMEL = 400;   // MEM_DIM*R
static constexpr int CKP  = 416;   // padded K for prenet1
static constexpr int CPRE = 256;
static constexpr int CL   = 512;   // LSTM
static constexpr int CG   = 2048;  // 4*LSTM
static constexpr int CATT = 128;
static constexpr int CF   = 32;    // conv filters
static constexpr int CKS  = 31;    // conv ksize
static constexpr int CTD  = 250;   // T_DEC
static constexpr int ROWSP = CTD*CB;   // 16000 prenet rows
static constexpr int ROWSA = CB*CT;    // 16384 (b,tau) rows

struct DecPtrs {
    // inputs (fp32 unless noted)
    const float *inputs, *memory; const unsigned char *mask;
    const float *pre_w1,*pre_b1,*pre_w2,*pre_b2;
    const float *l1_wih,*l1_whh,*l1_bih,*l1_bhh;
    const float *l2_wih,*l2_whh,*l2_bih,*l2_bhh;
    const float *aq_w,*aq_b,*aa_w,*aa_b,*conv_w,*al_w,*al_b,*v_w;
    const float *proj_w,*proj_b,*stop_w;
    // outputs
    float *out_frames,*out_attn,*out_stop;
    // f16 workspace
    _Float16 *w1ih,*w1hh,*w2ih,*w2hh,*wp1,*wp2,*wq,*wan,*wloc,*wproj;
    _Float16 *mem16,*in16,*P1,*P2,*x16,*h1_16,*h2_16,*pj16,*locf;
    // f32 workspace
    float *pannot,*ploc,*gates,*c1,*c2,*attn,*cum,*b1c,*b2c;
    int *bar;
};

__device__ inline v8f zero8() {
    v8f z;
#pragma unroll
    for (int i = 0; i < 8; ++i) z[i] = 0.f;
    return z;
}

__device__ inline float sigf(float x) { return 1.f / (1.f + __expf(-x)); }

#define PACK_A(a, lo, hi) _Pragma("unroll") \
    for (int _i = 0; _i < 8; ++_i) { (a)[_i] = (lo)[_i]; (a)[_i+8] = (hi)[_i]; }

// One 16x16 output tile: A row-major MxK f16 (lda), W row-major NxK f16 (ldb),
// C += A * W^T per CDNA5 WMMA lane layouts (ISA 7.12.2).
__device__ inline v8f wmma_acc(const _Float16* __restrict__ A, int lda, int m0,
                               const _Float16* __restrict__ W, int ldb, int n0,
                               int K, v8f acc) {
    const int lane = threadIdx.x & 31;
    const int hs = lane >> 4, l15 = lane & 15;
    const _Float16* ap = A + (size_t)(m0 + l15) * lda + hs * 8;
    const _Float16* bp = W + (size_t)(n0 + l15) * ldb + hs * 8;
    for (int kb = 0; kb < K; kb += 32) {
        v8h alo = *(const v8h*)(ap + kb);
        v8h ahi = *(const v8h*)(ap + kb + 16);
        v16h a, b;
        v8h blo = *(const v8h*)(bp + kb);
        v8h bhi = *(const v8h*)(bp + kb + 16);
        PACK_A(b, blo, bhi)
        PACK_A(a, alo, ahi)
        acc = __builtin_amdgcn_wmma_f32_16x16x32_f16(false, a, false, b,
                                                     (short)0, acc, false, false);
    }
    return acc;
}

// Two M-tiles (m0, m0+16) x one N-tile, dual A-matrix accumulate (gate GEMMs,
// lda=ldb=CL). B fragments are loaded once and reused for both M-tiles.
struct Acc2 { v8f a0, a1; };
__device__ inline Acc2 wmma_pair(const _Float16* __restrict__ A1,
                                 const _Float16* __restrict__ A2,
                                 const _Float16* __restrict__ B1,
                                 const _Float16* __restrict__ B2,
                                 int m0, int n0) {
    const int lane = threadIdx.x & 31;
    const int hs = lane >> 4, l15 = lane & 15;
    const _Float16* a1p = A1 + (size_t)(m0 + l15) * CL + hs * 8;
    const _Float16* a2p = A2 + (size_t)(m0 + l15) * CL + hs * 8;
    const _Float16* b1p = B1 + (size_t)(n0 + l15) * CL + hs * 8;
    const _Float16* b2p = B2 + (size_t)(n0 + l15) * CL + hs * 8;
    const int mstep = 16 * CL;
    Acc2 r; r.a0 = zero8(); r.a1 = zero8();
    for (int kb = 0; kb < CL; kb += 32) {
        v8h lo, hi; v16h a, b1, b2;
        lo = *(const v8h*)(b1p + kb); hi = *(const v8h*)(b1p + kb + 16);
        PACK_A(b1, lo, hi)
        lo = *(const v8h*)(b2p + kb); hi = *(const v8h*)(b2p + kb + 16);
        PACK_A(b2, lo, hi)
        lo = *(const v8h*)(a1p + kb); hi = *(const v8h*)(a1p + kb + 16);
        PACK_A(a, lo, hi)
        r.a0 = __builtin_amdgcn_wmma_f32_16x16x32_f16(false, a, false, b1, (short)0, r.a0, false, false);
        lo = *(const v8h*)(a1p + mstep + kb); hi = *(const v8h*)(a1p + mstep + kb + 16);
        PACK_A(a, lo, hi)
        r.a1 = __builtin_amdgcn_wmma_f32_16x16x32_f16(false, a, false, b1, (short)0, r.a1, false, false);
        lo = *(const v8h*)(a2p + kb); hi = *(const v8h*)(a2p + kb + 16);
        PACK_A(a, lo, hi)
        r.a0 = __builtin_amdgcn_wmma_f32_16x16x32_f16(false, a, false, b2, (short)0, r.a0, false, false);
        lo = *(const v8h*)(a2p + mstep + kb); hi = *(const v8h*)(a2p + mstep + kb + 16);
        PACK_A(a, lo, hi)
        r.a1 = __builtin_amdgcn_wmma_f32_16x16x32_f16(false, a, false, b2, (short)0, r.a1, false, false);
    }
    return r;
}

__device__ inline void st_f32(float* C, int ldc, int m0, int n0, v8f acc,
                              const float* bias) {
    const int lane = threadIdx.x & 31;
    const int hs = lane >> 4, l15 = lane & 15;
    const int n = n0 + l15;
    float bv = bias ? bias[n] : 0.f;
#pragma unroll
    for (int r = 0; r < 8; ++r)
        C[(size_t)(m0 + r + hs * 8) * ldc + n] = acc[r] + bv;
}

__device__ inline void st_f16_relu(_Float16* C, int ldc, int m0, int n0, v8f acc,
                                   const float* bias) {
    const int lane = threadIdx.x & 31;
    const int hs = lane >> 4, l15 = lane & 15;
    const int n = n0 + l15;
    float bv = bias[n];
#pragma unroll
    for (int r = 0; r < 8; ++r) {
        float v = acc[r] + bv;
        C[(size_t)(m0 + r + hs * 8) * ldc + n] = (_Float16)(v > 0.f ? v : 0.f);
    }
}

// Device-wide sense barrier (all NWG blocks resident by construction).
__device__ inline void gbar(int* bar, int& gen) {
    __threadfence();
    __syncthreads();
    if (threadIdx.x == 0) {
        int g = ++gen;
        int arrived = __hip_atomic_fetch_add(bar, 1, __ATOMIC_ACQ_REL,
                                             __HIP_MEMORY_SCOPE_AGENT);
        if (arrived == NWG - 1) {
            __hip_atomic_store(bar, 0, __ATOMIC_RELAXED, __HIP_MEMORY_SCOPE_AGENT);
            __hip_atomic_store(bar + 1, g, __ATOMIC_RELEASE, __HIP_MEMORY_SCOPE_AGENT);
        } else {
            while (__hip_atomic_load(bar + 1, __ATOMIC_ACQUIRE,
                                     __HIP_MEMORY_SCOPE_AGENT) < g)
                __builtin_amdgcn_s_sleep(2);
        }
    }
    __syncthreads();
    __threadfence();
}

__device__ inline void cvt(const float* s, _Float16* d, int n, int g) {
    for (int i = g; i < n; i += NTOT) d[i] = (_Float16)s[i];
}

// gates1 GEMM -> LDS tiles, then LSTM1 nonlinearity + location conv for step tn.
// WG w owns gate columns [16w,16w+16) of all 4 quadrants and batch rows {2w,2w+1}.
__device__ void prep_step(const DecPtrs& P, int tn,
                          float* s_g1, float* s_att, float* s_cum,
                          const float* s_convw) {
    const int wg = blockIdx.x, tid = threadIdx.x;
    const int wv = tid >> 5, lane = tid & 31;
    const int hs = lane >> 4, l15 = lane & 15;
    // ---- pre-sync: WMMA tiles into LDS + stage attn/cum rows ----
    {
        int q = wv & 3, mp = wv >> 2;                 // 8 waves: 2 m-pairs x 4 quadrants
        int n0 = (q * 32 + wg) * 16;                  // gate column base q*512 + 16*wg
        Acc2 r = wmma_pair(P.x16, P.h1_16, P.w1ih, P.w1hh, mp * 32, n0);
#pragma unroll
        for (int rt = 0; rt < 2; ++rt) {
            v8f acc = rt ? r.a1 : r.a0;
            int mb = mp * 32 + rt * 16;
#pragma unroll
            for (int rr = 0; rr < 8; ++rr)
                s_g1[(mb + rr + hs * 8) * 64 + q * 16 + l15] = acc[rr];
        }
    }
    for (int i = tid; i < 2 * CT; i += NTHR) {
        int rr = i >> 8, tau = i & 255;
        int b = 2 * wg + rr;
        s_att[i] = P.attn[(size_t)b * CT + tau];
        s_cum[i] = P.cum[(size_t)b * CT + tau];
    }
    __syncthreads();
    // ---- post-sync: LSTM1 elementwise from LDS ----
    for (int idx = tid; idx < 1024; idx += NTHR) {
        int b = idx >> 4, nl = idx & 15;
        int n = wg * 16 + nl;
        const float* gr = s_g1 + b * 64 + nl;
        float ig = gr[0]  + P.b1c[n];
        float fg = gr[16] + P.b1c[CL + n];
        float gg = gr[32] + P.b1c[2 * CL + n];
        float og = gr[48] + P.b1c[3 * CL + n];
        float cn = sigf(fg) * P.c1[(size_t)b * CL + n] + sigf(ig) * tanhf(gg);
        P.c1[(size_t)b * CL + n] = cn;
        P.h1_16[(size_t)b * CL + n] = (_Float16)(sigf(og) * tanhf(cn));
    }
    // ---- location conv for owned rows (attn/cum from LDS) ----
    for (int rr = 0; rr < 2; ++rr) {
        int b = 2 * wg + rr, tau = tid;
        float accf[CF];
#pragma unroll
        for (int f = 0; f < CF; ++f) accf[f] = 0.f;
        for (int j = 0; j < CKS; ++j) {
            int tj = tau + j - (CKS - 1) / 2;
            if (tj >= 0 && tj < CT) {
                float a = s_att[rr * CT + tj];
                float c = s_cum[rr * CT + tj];
#pragma unroll
                for (int f = 0; f < CF; ++f)
                    accf[f] += s_convw[f * 62 + j] * a + s_convw[f * 62 + CKS + j] * c;
            }
        }
        size_t p = (size_t)b * CT + tau;
#pragma unroll
        for (int cc = 0; cc < 4; ++cc) {
            v8h pk;
#pragma unroll
            for (int e = 0; e < 8; ++e) pk[e] = (_Float16)accf[cc * 8 + e];
            *(v8h*)(P.locf + p * CF + cc * 8) = pk;
        }
    }
}

__global__ void bar_init_kernel(int* bar) { bar[0] = 0; bar[1] = 0; }

__global__ void __launch_bounds__(NTHR, 1) decoder_persistent(DecPtrs P) {
    __shared__ float s_convw[CF * 2 * CKS]; // 1984 floats
    __shared__ float s_v[CATT];
    __shared__ float s_red[NTHR];
    __shared__ float s_g1[64 * 64];         // gates1 tiles [b][q*16+nl], 16KB
    __shared__ float s_h2[2 * CL];          // owned h2 rows
    __shared__ float s_q[2 * CATT];
    __shared__ float s_align[2 * CT];
    __shared__ float s_att[2 * CT];
    __shared__ float s_cum[2 * CT];

    const int g     = blockIdx.x * NTHR + threadIdx.x;
    const int wg    = blockIdx.x;
    const int tid   = threadIdx.x;
    const int gwave = blockIdx.x * (NTHR / 32) + (tid >> 5);
    int gen = 0;

    for (int i = tid; i < CF * 2 * CKS; i += NTHR) s_convw[i] = P.conv_w[i];
    for (int i = tid; i < CATT; i += NTHR) s_v[i] = P.v_w[i];
    __syncthreads();

    // ---------------- P0a: conversions + combined biases + state init ----------------
    cvt(P.l1_wih, P.w1ih, CG * CL, g);
    cvt(P.l1_whh, P.w1hh, CG * CL, g);
    cvt(P.l2_wih, P.w2ih, CG * CL, g);
    cvt(P.l2_whh, P.w2hh, CG * CL, g);
    cvt(P.pre_w2, P.wp2, CPRE * CPRE, g);
    cvt(P.aq_w,   P.wq,   CATT * CL, g);
    cvt(P.aa_w,   P.wan,  CATT * CIN, g);
    cvt(P.al_w,   P.wloc, CATT * CF, g);
    cvt(P.proj_w, P.wproj, CMEL * 768, g);
    cvt(P.inputs, P.in16, ROWSA * CIN, g);
    for (int i = g; i < CG; i += NTOT) {
        P.b1c[i] = P.l1_bih[i] + P.l1_bhh[i];
        P.b2c[i] = P.l2_bih[i] + P.l2_bhh[i];
    }
    for (int i = g; i < CPRE * CKP; i += NTOT) {
        int r = i / CKP, c = i % CKP;
        P.wp1[i] = (_Float16)((c < CMEL) ? P.pre_w1[(size_t)r * CMEL + c] : 0.f);
    }
    for (int i = g; i < ROWSP * CKP; i += NTOT) { // teacher-forced, t-1 shifted
        int row = i / CKP, c = i % CKP;
        int tt = row / CB, b = row % CB;
        float v = 0.f;
        if (c < CMEL && tt > 0)
            v = P.memory[(size_t)b * (CTD * CMEL) + (size_t)(tt - 1) * CMEL + c];
        P.mem16[i] = (_Float16)v;
    }
    for (int i = g; i < CB * CL; i += NTOT) {
        P.c1[i] = 0.f; P.c2[i] = 0.f;
        P.x16[i] = (_Float16)0.f; P.h1_16[i] = (_Float16)0.f; P.h2_16[i] = (_Float16)0.f;
    }
    for (int i = g; i < CB * CT; i += NTOT) { P.attn[i] = 0.f; P.cum[i] = 0.f; }
    gbar(P.bar, gen);

    // ---------------- P0b: prenet L1 (all steps) + proc_annot ----------------
    for (int tile = gwave; tile < (ROWSP / 16) * (CPRE / 16); tile += NWAVE) {
        int mt = tile / 16, nt = tile % 16;
        v8f acc = wmma_acc(P.mem16, CKP, mt * 16, P.wp1, CKP, nt * 16, CKP, zero8());
        st_f16_relu(P.P1, CPRE, mt * 16, nt * 16, acc, P.pre_b1);
    }
    for (int tile = gwave; tile < (ROWSA / 16) * (CATT / 16); tile += NWAVE) {
        int mt = tile / 8, nt = tile % 8;
        v8f acc = wmma_acc(P.in16, CIN, mt * 16, P.wan, CIN, nt * 16, CIN, zero8());
        st_f32(P.pannot, CATT, mt * 16, nt * 16, acc, P.aa_b);
    }
    gbar(P.bar, gen);

    // ---------------- P0c: prenet L2 (all steps) ----------------
    for (int tile = gwave; tile < (ROWSP / 16) * (CPRE / 16); tile += NWAVE) {
        int mt = tile / 16, nt = tile % 16;
        v8f acc = wmma_acc(P.P1, CPRE, mt * 16, P.wp2, CPRE, nt * 16, CPRE, zero8());
        st_f16_relu(P.P2, CPRE, mt * 16, nt * 16, acc, P.pre_b2);
    }
    gbar(P.bar, gen);

    // ---------------- P0d: assemble x for step 0 ----------------
    for (int i = g; i < CB * CPRE; i += NTOT) {
        int b = i / CPRE, j = i % CPRE;
        P.x16[(size_t)b * CL + j] = P.P2[(size_t)b * CPRE + j];
    }
    gbar(P.bar, gen);

    // ---------------- P0e: gates1 + LSTM1 + conv for t=0 ----------------
    prep_step(P, 0, s_g1, s_att, s_cum, s_convw);
    gbar(P.bar, gen);

    // ================= 250-step scan, 3 device barriers per step =================
    for (int t = 0; t < CTD; ++t) {
        // --- T2: gates2 GEMM (dual-tile) + loc-linear GEMM ---
        {
            int mp = gwave >> 7, nt = gwave & 127;
            Acc2 r = wmma_pair(P.h1_16, P.h2_16, P.w2ih, P.w2hh, mp * 32, nt * 16);
            st_f32(P.gates, CG, mp * 32,      nt * 16, r.a0, nullptr);
            st_f32(P.gates, CG, mp * 32 + 16, nt * 16, r.a1, nullptr);
        }
        for (int tile = gwave; tile < (ROWSA / 16) * (CATT / 16); tile += NWAVE) {
            int mt = tile / 8, nt = tile % 8;
            v8f acc = wmma_acc(P.locf, CF, mt * 16, P.wloc, CF, nt * 16, CF, zero8());
            st_f32(P.ploc, CATT, mt * 16, nt * 16, acc, P.al_b);
        }
        gbar(P.bar, gen);

        // --- T3: LSTM2 + proc_q + align + softmax + ctx (WG-owned rows 2wg,2wg+1) ---
        for (int idx = tid; idx < 2 * CL; idx += NTHR) {
            int rr = idx >> 9, n = idx & (CL - 1);
            int b = 2 * wg + rr;
            const float* gp = P.gates + (size_t)b * CG;
            float ig = gp[n]          + P.b2c[n];
            float fg = gp[CL + n]     + P.b2c[CL + n];
            float gg = gp[2 * CL + n] + P.b2c[2 * CL + n];
            float og = gp[3 * CL + n] + P.b2c[3 * CL + n];
            float cn = sigf(fg) * P.c2[(size_t)b * CL + n] + sigf(ig) * tanhf(gg);
            P.c2[(size_t)b * CL + n] = cn;
            float h = sigf(og) * tanhf(cn);
            s_h2[idx] = h;
            _Float16 hh = (_Float16)h;
            P.h2_16[(size_t)b * CL + n] = hh;
            P.pj16[(size_t)b * 768 + n] = hh;
        }
        __syncthreads();
        { // proc_q for owned rows: thread -> (rr, k)
            int rr = tid >> 7, k = tid & (CATT - 1);
            float q = P.aq_b[k];
            const v8h* wr = (const v8h*)(P.wq + (size_t)k * CL);
            const float* hr = s_h2 + rr * CL;
            for (int j = 0; j < CL / 8; ++j) {
                v8h wv = wr[j];
#pragma unroll
                for (int e = 0; e < 8; ++e) q += hr[j * 8 + e] * (float)wv[e];
            }
            s_q[rr * CATT + k] = q;
        }
        __syncthreads();
        for (int rr = 0; rr < 2; ++rr) { // align
            int b = 2 * wg + rr, tau = tid;
            size_t p = (size_t)b * CT + tau;
            const float4* a4 = (const float4*)(P.pannot + p * CATT);
            const float4* l4 = (const float4*)(P.ploc + p * CATT);
            const float* qr = s_q + rr * CATT;
            float s = 0.f;
            for (int k4 = 0; k4 < CATT / 4; ++k4) {
                float4 av = a4[k4], lv = l4[k4];
                int k = k4 * 4;
                s += tanhf(qr[k]     + av.x + lv.x) * s_v[k]
                   + tanhf(qr[k + 1] + av.y + lv.y) * s_v[k + 1]
                   + tanhf(qr[k + 2] + av.z + lv.z) * s_v[k + 2]
                   + tanhf(qr[k + 3] + av.w + lv.w) * s_v[k + 3];
            }
            s_align[rr * CT + tau] = P.mask[p] ? s : -1e9f;
        }
        __syncthreads();
        for (int rr = 0; rr < 2; ++rr) { // softmax + ctx + next-step operand assembly
            int b = 2 * wg + rr;
            float v = s_align[rr * CT + tid];
            s_red[tid] = v; __syncthreads();
            for (int o = 128; o > 0; o >>= 1) {
                if (tid < o) s_red[tid] = fmaxf(s_red[tid], s_red[tid + o]);
                __syncthreads();
            }
            float mx = s_red[0]; __syncthreads();
            float e = __expf(v - mx);
            s_red[tid] = e; __syncthreads();
            for (int o = 128; o > 0; o >>= 1) {
                if (tid < o) s_red[tid] += s_red[tid + o];
                __syncthreads();
            }
            float inv = 1.f / s_red[0]; __syncthreads();
            float a = e * inv;
            P.attn[(size_t)b * CT + tid] = a;
            P.cum[(size_t)b * CT + tid] += a;
            P.out_attn[(size_t)b * (CTD * CT) + (size_t)t * CT + tid] = a;
            s_red[tid] = a; __syncthreads();
            float ctx = 0.f;
            const float* inp = P.inputs + ((size_t)b * CT) * CIN + tid;
            for (int tau = 0; tau < CT; ++tau) ctx += s_red[tau] * inp[(size_t)tau * CIN];
            _Float16 ch = (_Float16)ctx;
            P.x16[(size_t)b * CL + CPRE + tid] = ch;
            P.pj16[(size_t)b * 768 + CL + tid] = ch;
            if (t + 1 < CTD)
                P.x16[(size_t)b * CL + tid] =
                    P.P2[((size_t)(t + 1) * CB + b) * CPRE + tid];
            __syncthreads();
        }
        gbar(P.bar, gen);

        // --- T4: proj + stop + (gates1 + LSTM1 + conv for step t+1) ---
        for (int tile = gwave; tile < 4 * 25; tile += NWAVE) {
            int mt = tile / 25, nt = tile % 25;
            v8f acc = wmma_acc(P.pj16, 768, mt * 16, P.wproj, 768, nt * 16, 768, zero8());
            const int lane = tid & 31;
            const int hs = lane >> 4, l15 = lane & 15;
            const int n = nt * 16 + l15;
            float bv = P.proj_b[n];
#pragma unroll
            for (int r = 0; r < 8; ++r) {
                int b = mt * 16 + r + hs * 8;
                P.out_frames[(size_t)b * (CTD * CMEL) + (size_t)t * CMEL + n] = acc[r] + bv;
            }
        }
        for (int i = g; i < CB; i += NTOT) { // stop head
            const v8h* pj = (const v8h*)(P.pj16 + (size_t)i * 768);
            float s = 0.f;
            for (int j = 0; j < 768 / 8; ++j) {
                v8h pv = pj[j];
#pragma unroll
                for (int e = 0; e < 8; ++e) s += (float)pv[e] * P.stop_w[j * 8 + e];
            }
            P.out_stop[(size_t)i * CTD + t] = 1.f / (1.f + __expf(-s));
        }
        if (t + 1 < CTD)
            prep_step(P, t + 1, s_g1, s_att, s_cum, s_convw);
        gbar(P.bar, gen);
    }
}

extern "C" void kernel_launch(void* const* d_in, const int* in_sizes, int n_in,
                              void* d_out, int out_size, void* d_ws, size_t ws_size,
                              hipStream_t stream) {
    (void)in_sizes; (void)n_in; (void)out_size; (void)ws_size;
    DecPtrs P;
    P.inputs  = (const float*)d_in[0];
    P.memory  = (const float*)d_in[1];
    P.mask    = (const unsigned char*)d_in[2];
    P.pre_w1  = (const float*)d_in[3];  P.pre_b1 = (const float*)d_in[4];
    P.pre_w2  = (const float*)d_in[5];  P.pre_b2 = (const float*)d_in[6];
    P.l1_wih  = (const float*)d_in[7];  P.l1_whh = (const float*)d_in[8];
    P.l1_bih  = (const float*)d_in[9];  P.l1_bhh = (const float*)d_in[10];
    P.l2_wih  = (const float*)d_in[11]; P.l2_whh = (const float*)d_in[12];
    P.l2_bih  = (const float*)d_in[13]; P.l2_bhh = (const float*)d_in[14];
    P.aq_w    = (const float*)d_in[15]; P.aq_b   = (const float*)d_in[16];
    P.aa_w    = (const float*)d_in[17]; P.aa_b   = (const float*)d_in[18];
    P.conv_w  = (const float*)d_in[19];
    P.al_w    = (const float*)d_in[20]; P.al_b   = (const float*)d_in[21];
    P.v_w     = (const float*)d_in[22];
    P.proj_w  = (const float*)d_in[23]; P.proj_b = (const float*)d_in[24];
    P.stop_w  = (const float*)d_in[25];

    P.out_frames = (float*)d_out;
    P.out_attn   = P.out_frames + (size_t)CB * CTD * CMEL;   // +6,400,000
    P.out_stop   = P.out_attn + (size_t)CB * CTD * CT;       // +4,096,000

    char* base = (char*)d_ws; size_t off = 0;
    auto alloc = [&](size_t bytes) -> char* {
        char* p = base + off; off = (off + bytes + 255) & ~(size_t)255; return p;
    };
    P.bar   = (int*)alloc(256);
    P.w1ih  = (_Float16*)alloc((size_t)CG * CL * 2);
    P.w1hh  = (_Float16*)alloc((size_t)CG * CL * 2);
    P.w2ih  = (_Float16*)alloc((size_t)CG * CL * 2);
    P.w2hh  = (_Float16*)alloc((size_t)CG * CL * 2);
    P.wp1   = (_Float16*)alloc((size_t)CPRE * CKP * 2);
    P.wp2   = (_Float16*)alloc((size_t)CPRE * CPRE * 2);
    P.wq    = (_Float16*)alloc((size_t)CATT * CL * 2);
    P.wan   = (_Float16*)alloc((size_t)CATT * CIN * 2);
    P.wloc  = (_Float16*)alloc((size_t)CATT * CF * 2);
    P.wproj = (_Float16*)alloc((size_t)CMEL * 768 * 2);
    P.mem16 = (_Float16*)alloc((size_t)ROWSP * CKP * 2);
    P.in16  = (_Float16*)alloc((size_t)ROWSA * CIN * 2);
    P.P1    = (_Float16*)alloc((size_t)ROWSP * CPRE * 2);
    P.P2    = (_Float16*)alloc((size_t)ROWSP * CPRE * 2);
    P.x16   = (_Float16*)alloc((size_t)CB * CL * 2);
    P.h1_16 = (_Float16*)alloc((size_t)CB * CL * 2);
    P.h2_16 = (_Float16*)alloc((size_t)CB * CL * 2);
    P.pj16  = (_Float16*)alloc((size_t)CB * 768 * 2);
    P.locf  = (_Float16*)alloc((size_t)ROWSA * CF * 2);
    P.pannot = (float*)alloc((size_t)ROWSA * CATT * 4);
    P.ploc   = (float*)alloc((size_t)ROWSA * CATT * 4);
    P.gates  = (float*)alloc((size_t)CB * CG * 4);
    P.c1     = (float*)alloc((size_t)CB * CL * 4);
    P.c2     = (float*)alloc((size_t)CB * CL * 4);
    P.attn   = (float*)alloc((size_t)CB * CT * 4);
    P.cum    = (float*)alloc((size_t)CB * CT * 4);
    P.b1c    = (float*)alloc((size_t)CG * 4);
    P.b2c    = (float*)alloc((size_t)CG * 4);

    bar_init_kernel<<<1, 1, 0, stream>>>(P.bar);
    decoder_persistent<<<NWG, NTHR, 0, stream>>>(P);
}
